// EMACodebook_42460046688449
// MI455X (gfx1250) — matrix operand order repaired and checked by reading, not compiled, not run on hardware
//
#include <hip/hip_runtime.h>

// ---------------------------------------------------------------------------
// EMA codebook / VQ clustering for MI455X (gfx1250, wave32, WMMA)
// B=4, L=4096, C=1024, K=4096  ->  N = 16384 tokens
// ---------------------------------------------------------------------------

#define N_TOK   16384
#define CDIM    1024
#define KCODES  4096
#define MOM     0.99f
#define OMOM    0.01f
#define EPS     1e-6f

typedef __attribute__((ext_vector_type(16))) __bf16 v16bf;
typedef __attribute__((ext_vector_type(8)))  float  v8f;

struct Pack32 { uint4 a, b; };   // 32 bytes == 16 bf16

__device__ __forceinline__ unsigned short f32_to_bf16(float f) {
    unsigned int u = __float_as_uint(f);
    unsigned int r = u + 0x7FFFu + ((u >> 16) & 1u);   // round-to-nearest-even
    return (unsigned short)(r >> 16);
}

__device__ __forceinline__ float blockReduceSum(float v, float* sh) {
    __syncthreads();                       // allow repeated use of sh
    #pragma unroll
    for (int o = 16; o > 0; o >>= 1) v += __shfl_xor(v, o, 32);
    int lane = threadIdx.x & 31, wid = threadIdx.x >> 5;
    if (lane == 0) sh[wid] = v;
    __syncthreads();
    int nw = blockDim.x >> 5;
    v = (threadIdx.x < nw) ? sh[threadIdx.x] : 0.0f;
    if (wid == 0) {
        #pragma unroll
        for (int o = 16; o > 0; o >>= 1) v += __shfl_xor(v, o, 32);
    }
    if (threadIdx.x == 0) sh[0] = v;
    __syncthreads();
    return sh[0];
}

// ---------------------------------------------------------------- zero scratch
__global__ void vq_zero(float* p, unsigned long long n) {
    unsigned long long i = (unsigned long long)blockIdx.x * blockDim.x + threadIdx.x;
    unsigned long long stride = (unsigned long long)gridDim.x * blockDim.x;
    for (; i < n; i += stride) p[i] = 0.0f;
}

// ------------------------------------------- normalize tokens -> bf16 rows
__global__ void vq_norm_feat(const float* __restrict__ x,
                             unsigned short* __restrict__ featbf) {
    __shared__ float sh[32];
    int row = blockIdx.x, tid = threadIdx.x;
    const float4 v = ((const float4*)(x + (size_t)row * CDIM))[tid];
    float s = v.x * v.x + v.y * v.y + v.z * v.z + v.w * v.w;
    s = blockReduceSum(s, sh);
    float inv = 1.0f / fmaxf(sqrtf(s), EPS);
    unsigned int p0 = (unsigned int)f32_to_bf16(v.x * inv) |
                      ((unsigned int)f32_to_bf16(v.y * inv) << 16);
    unsigned int p1 = (unsigned int)f32_to_bf16(v.z * inv) |
                      ((unsigned int)f32_to_bf16(v.w * inv) << 16);
    uint2 p; p.x = p0; p.y = p1;
    ((uint2*)(featbf + (size_t)row * CDIM))[tid] = p;
}

// -------------------------------- codebook -> bf16 + squared-norm per code
__global__ void vq_prep_cb(const float* __restrict__ cb,
                           unsigned short* __restrict__ cbbf,
                           float* __restrict__ c2) {
    __shared__ float sh[32];
    int k = blockIdx.x, tid = threadIdx.x;
    const float4 v = ((const float4*)(cb + (size_t)k * CDIM))[tid];
    float s = v.x * v.x + v.y * v.y + v.z * v.z + v.w * v.w;
    s = blockReduceSum(s, sh);
    if (tid == 0) c2[k] = s;
    unsigned int p0 = (unsigned int)f32_to_bf16(v.x) |
                      ((unsigned int)f32_to_bf16(v.y) << 16);
    unsigned int p1 = (unsigned int)f32_to_bf16(v.z) |
                      ((unsigned int)f32_to_bf16(v.w) << 16);
    uint2 p; p.x = p0; p.y = p1;
    ((uint2*)(cbbf + (size_t)k * CDIM))[tid] = p;
}

// ------------------------------------------------ fused GEMM + argmin kernel
// block = 128 thr (4 waves). Block owns 64 token rows, streams K in chunks of
// 32 codes (double-buffered in LDS). Wave (wr,wc) = (w>>1, w&1) computes a
// 32-row x 16-code tile: 2 accumulators, 1 B fragment per 2 WMMAs.
#define LDA 1032                      // 1024 + 8 bf16 pad (keeps 16B align)
#define KCHUNK 32
#define NCHUNK (KCODES / KCHUNK)      // 128

__device__ __forceinline__ Pack32 ldsFragA(const char* rowk, int cc) {
    Pack32 r;
    const char* p = rowk + cc * 64;
    r.a = *(const uint4*)(p);
    r.b = *(const uint4*)(p + 32);
    return r;
}
__device__ __forceinline__ Pack32 ldsFragB(const char* rowk, int cc) {
    Pack32 r;
    const char* p = rowk + cc * 64;
    r.a = *(const uint4*)(p);
    r.b = *(const uint4*)(p + 32);
    return r;
}

__global__ void __launch_bounds__(128, 1)
vq_argmin(const unsigned short* __restrict__ featbf,
          const unsigned short* __restrict__ cbbf,
          const float* __restrict__ c2,
          float* __restrict__ ids) {
    extern __shared__ unsigned short smem[];
    unsigned short* At  = smem;                            // 64 x LDA bf16
    char* Bt0 = (char*)(smem + 64 * LDA);                  // 32 x LDA bf16
    char* Bt1 = Bt0 + KCHUNK * LDA * 2;                    // 32 x LDA bf16
    __shared__ float s_bv[4][32];
    __shared__ int   s_bi[4][32];

    const int rowBase = blockIdx.x * 64;
    const int tid  = threadIdx.x;
    const int wave = tid >> 5;
    const int lane = tid & 31;
    const int m    = lane & 15;       // row (A) / col (B) within 16-tile
    const int khi  = lane >> 4;       // K-half selector
    const int wr   = wave >> 1;       // row half of block (32 rows)
    const int wc   = wave & 1;        // code half of chunk (16 codes)

    // ---- stage A tile (64 rows x 1024 bf16) and first B chunk ----
    for (int idx = tid; idx < 64 * 128; idx += 128) {
        int r = idx >> 7, c16 = idx & 127;
        uint4 v = ((const uint4*)(featbf + (size_t)(rowBase + r) * CDIM))[c16];
        *(uint4*)((char*)At + ((size_t)r * LDA + c16 * 8) * 2) = v;
    }
    for (int idx = tid; idx < KCHUNK * 128; idx += 128) {
        int r = idx >> 7, c16 = idx & 127;
        uint4 v = ((const uint4*)(cbbf + (size_t)r * CDIM))[c16];
        *(uint4*)(Bt0 + ((size_t)r * LDA + c16 * 8) * 2) = v;
    }
    __syncthreads();

    float bestVal[16];
    int   bestIdx[16];
    #pragma unroll
    for (int i = 0; i < 16; ++i) { bestVal[i] = 3.4e38f; bestIdx[i] = 0; }

    // per-lane fragment row pointers (khi offset folded in)
    const char* a0rowk = (const char*)At +
        ((size_t)(wr * 32 + m) * LDA) * 2 + khi * 16;
    const char* a1rowk = a0rowk + (size_t)16 * LDA * 2;
    const size_t browOff = ((size_t)(wc * 16 + m) * LDA) * 2 + khi * 16;

    for (int kbi = 0; kbi < NCHUNK; ++kbi) {
        const int   kb    = kbi * KCHUNK;
        const char* bbase = (kbi & 1) ? Bt1 : Bt0;
        char*       bnext = (kbi & 1) ? Bt0 : Bt1;
        const bool  hasNext = (kbi + 1) < NCHUNK;
        const int   nkb   = kb + KCHUNK;
        const char* browk = bbase + browOff;

        v8f acc0 = {}, acc1 = {};
        Pack32 pa0 = ldsFragA(a0rowk, 0);
        Pack32 pa1 = ldsFragA(a1rowk, 0);
        Pack32 pb  = ldsFragB(browk, 0);

        #pragma unroll
        for (int q = 0; q < 8; ++q) {
            // issue global prefetch of next chunk, stage q (4 x b128)
            // named scalars (NOT an array) so they are guaranteed in VGPRs
            uint4 p0, p1, p2, p3;
            if (hasNext) {
                const uint4* src = (const uint4*)(cbbf +
                    (size_t)(nkb + q * 4) * CDIM) + tid;
                p0 = src[0 * (CDIM / 8)];
                p1 = src[1 * (CDIM / 8)];
                p2 = src[2 * (CDIM / 8)];
                p3 = src[3 * (CDIM / 8)];
            }
            // 4 cc steps x 2 WMMA, fragments rotated one step ahead
            #pragma unroll
            for (int s = 0; s < 4; ++s) {
                int cc = q * 4 + s;
                int nx = (cc + 1) & 31;
                Pack32 na0 = ldsFragA(a0rowk, nx);
                Pack32 na1 = ldsFragA(a1rowk, nx);
                Pack32 nb  = ldsFragB(browk, nx);
                v16bf av0 = __builtin_bit_cast(v16bf, pa0);
                v16bf av1 = __builtin_bit_cast(v16bf, pa1);
                v16bf bv  = __builtin_bit_cast(v16bf, pb);
                acc0 = __builtin_amdgcn_wmma_f32_16x16x32_bf16(
                    false, av0, false, bv, (short)0, acc0, false, false);
                acc1 = __builtin_amdgcn_wmma_f32_16x16x32_bf16(
                    false, av1, false, bv, (short)0, acc1, false, false);
                pa0 = na0; pa1 = na1; pb = nb;
            }
            // park prefetched data in the other B buffer
            if (hasNext) {
                char* dst = bnext + ((size_t)(q * 4) * LDA + tid * 8) * 2;
                *(uint4*)(dst + 0 * LDA * 2) = p0;
                *(uint4*)(dst + 1 * LDA * 2) = p1;
                *(uint4*)(dst + 2 * LDA * 2) = p2;
                *(uint4*)(dst + 3 * LDA * 2) = p3;
            }
        }

        // ---- per-lane running argmin (no cross-lane work in hot loop) ----
        float c2n  = c2[kb + wc * 16 + m];
        int   code = kb + wc * 16 + m;
        #pragma unroll
        for (int i = 0; i < 8; ++i) {
            float v0 = fmaf(-2.0f, acc0[i], c2n);
            bool  t0 = v0 < bestVal[i];
            bestVal[i] = t0 ? v0 : bestVal[i];
            bestIdx[i] = t0 ? code : bestIdx[i];
            float v1 = fmaf(-2.0f, acc1[i], c2n);
            bool  t1 = v1 < bestVal[8 + i];
            bestVal[8 + i] = t1 ? v1 : bestVal[8 + i];
            bestIdx[8 + i] = t1 ? code : bestIdx[8 + i];
        }
        __syncthreads();   // buffer swap barrier
    }

    // ---- epilogue: butterfly across 16-lane halves, then cross-wave merge ----
    #pragma unroll
    for (int j = 0; j < 2; ++j) {
        #pragma unroll
        for (int i = 0; i < 8; ++i) {
            float val = bestVal[j * 8 + i];
            int   idx = bestIdx[j * 8 + i];
            #pragma unroll
            for (int off = 1; off < 16; off <<= 1) {
                float ov = __shfl_xor(val, off, 32);
                int   oi = __shfl_xor(idx, off, 32);
                bool  t = (ov < val) || (ov == val && oi < idx);
                val = t ? ov : val;
                idx = t ? oi : idx;
            }
            int rloc = j * 16 + khi * 8 + i;   // row within wave's 32
            if (m == 0) { s_bv[wave][rloc] = val; s_bi[wave][rloc] = idx; }
        }
    }
    __syncthreads();
    if (tid < 64) {
        int half = tid >> 5, r = tid & 31;
        float v0 = s_bv[half * 2][r],     v1 = s_bv[half * 2 + 1][r];
        int   i0 = s_bi[half * 2][r],     i1 = s_bi[half * 2 + 1][r];
        bool  t = (v1 < v0) || (v1 == v0 && i1 < i0);
        ids[rowBase + half * 32 + r] = (float)(t ? i1 : i0);
    }
}

// ---------------------------------------------- scatter: sums[id] += feat
__global__ void vq_scatter(const float* __restrict__ x,
                           const float* __restrict__ ids,
                           float* __restrict__ sums,
                           float* __restrict__ counts) {
    __shared__ float sh[32];
    int row = blockIdx.x, tid = threadIdx.x;
    const float4 v = ((const float4*)(x + (size_t)row * CDIM))[tid];
    float s = v.x * v.x + v.y * v.y + v.z * v.z + v.w * v.w;
    s = blockReduceSum(s, sh);
    float inv = 1.0f / fmaxf(sqrtf(s), EPS);
    int id = (int)ids[row];
    float* srow = sums + (size_t)id * CDIM + tid * 4;
    atomicAdd(srow + 0, v.x * inv);
    atomicAdd(srow + 1, v.y * inv);
    atomicAdd(srow + 2, v.z * inv);
    atomicAdd(srow + 3, v.w * inv);
    if (tid == 0) atomicAdd(&counts[id], 1.0f);
}

// ---------------------------------------------- EMA update + renormalize
__global__ void vq_finalize(const float* __restrict__ cb,
                            const float* __restrict__ sums,
                            const float* __restrict__ counts,
                            float* __restrict__ newcb) {
    __shared__ float sh[32];
    int k = blockIdx.x, tid = threadIdx.x;
    float cnt = counts[k];
    const float4 cv = ((const float4*)(cb   + (size_t)k * CDIM))[tid];
    const float4 sv = ((const float4*)(sums + (size_t)k * CDIM))[tid];
    float ic = 1.0f / fmaxf(cnt, 1.0f);
    float m0 = sv.x * ic, m1 = sv.y * ic, m2 = sv.z * ic, m3 = sv.w * ic;
    float n1 = blockReduceSum(m0 * m0 + m1 * m1 + m2 * m2 + m3 * m3, sh);
    float i1 = 1.0f / fmaxf(sqrtf(n1), EPS);
    float u0 = MOM * cv.x + OMOM * (m0 * i1);
    float u1 = MOM * cv.y + OMOM * (m1 * i1);
    float u2 = MOM * cv.z + OMOM * (m2 * i1);
    float u3 = MOM * cv.w + OMOM * (m3 * i1);
    float n2 = blockReduceSum(u0 * u0 + u1 * u1 + u2 * u2 + u3 * u3, sh);
    float i2 = 1.0f / fmaxf(sqrtf(n2), EPS);
    bool used = cnt > 0.0f;
    float4 o;
    o.x = used ? u0 * i2 : cv.x;
    o.y = used ? u1 * i2 : cv.y;
    o.z = used ? u2 * i2 : cv.z;
    o.w = used ? u3 * i2 : cv.w;
    ((float4*)(newcb + (size_t)k * CDIM))[tid] = o;
}

// ---------------------------------------------------------------------------
extern "C" void kernel_launch(void* const* d_in, const int* in_sizes, int n_in,
                              void* d_out, int out_size, void* d_ws, size_t ws_size,
                              hipStream_t stream) {
    const float* x  = (const float*)d_in[0];   // (B,L,C) = N_TOK x CDIM
    const float* cb = (const float*)d_in[1];   // (K, C)
    float* out = (float*)d_out;                // [ids (N)] ++ [new_cb (K*C)]

    // scratch layout
    unsigned short* featbf = (unsigned short*)d_ws;              // N*C bf16
    unsigned short* cbbf   = featbf + (size_t)N_TOK * CDIM;      // K*C bf16
    float* c2     = (float*)(cbbf + (size_t)KCODES * CDIM);      // K
    float* sums   = c2 + KCODES;                                 // K*C
    float* counts = sums + (size_t)KCODES * CDIM;                // K

    // 1. zero the scatter accumulators (sums ++ counts are contiguous)
    vq_zero<<<1024, 256, 0, stream>>>(sums,
        (unsigned long long)KCODES * CDIM + KCODES);
    // 2. normalize tokens -> bf16
    vq_norm_feat<<<N_TOK, 256, 0, stream>>>(x, featbf);
    // 3. codebook -> bf16 + |c|^2
    vq_prep_cb<<<KCODES, 256, 0, stream>>>(cb, cbbf, c2);
    // 4. fused WMMA GEMM + argmin: 64 rows/block, LDS = A + 2x B chunk
    size_t ldsBytes = (size_t)(64 + 2 * KCHUNK) * LDA * 2;
    vq_argmin<<<N_TOK / 64, 128, ldsBytes, stream>>>(featbf, cbbf, c2, out);
    // 5. scatter-reduce per-cluster sums/counts
    vq_scatter<<<N_TOK, 256, 0, stream>>>(x, out, sums, counts);
    // 6. EMA update + renormalize codebook
    vq_finalize<<<KCODES, 256, 0, stream>>>(cb, sums, counts, out + N_TOK);
}
